// fNIRS_T_39058432590447
// MI455X (gfx1250) — matrix operand order repaired and checked by reading, not compiled, not run on hardware
//
#include <hip/hip_runtime.h>
#include <hip/hip_bf16.h>
#include <math.h>

typedef __attribute__((ext_vector_type(16))) _Float16 v16h;
typedef __attribute__((ext_vector_type(8)))  _Float16 v8h;
typedef __attribute__((ext_vector_type(4)))  _Float16 v4h;
typedef __attribute__((ext_vector_type(8)))  float    v8f;

#define N_CH   53
#define N_REG  7
#define N_TOK  60
#define SPLEN  512
#define NFREQ  20
#define DM     512
#define FD     1024
#define HEADS  8
#define DH     64
#define MLPD   4096

// ---- compile-time 60x60 attention mask as 64-bit row bitmaps ---------------
struct MaskTab { unsigned long long row[64]; };
static constexpr MaskTab make_mask() {
  MaskTab t = {};
  int reg_of[54] = { -1,
    0,2,2,0,2,4,2,2,6,0, 4,5,2,4,6,6,4,4,6,4,
    6,6,6,5,4,5,6,6,6,6, 4,4,6,4,6,6,6,5,4,1,
    6,4,6,3,4,3,1,6,3,3, 3,1,3 };
  for (int i = 0; i < 64; ++i) {
    unsigned long long m = 0;
    if (i < 60) {
      for (int j = 0; j < 60; ++j) {
        bool ri = (i < 7), rj = (j < 7);
        bool ok;
        if (ri == rj) ok = true;
        else if (ri)  ok = (reg_of[j - 6] == i);
        else          ok = (reg_of[i - 6] == j);
        if (ok) m |= 1ull << j;
      }
    }
    t.row[i] = m;
  }
  return t;
}
__device__ __constant__ MaskTab MASK = make_mask();

__device__ inline float gelu_exact(float x) {
  return 0.5f * x * (1.0f + erff(x * 0.70710678118654752440f));
}

__device__ inline v16h cat8(v8h lo, v8h hi) {
  return __builtin_shufflevector(lo, hi, 0,1,2,3,4,5,6,7,8,9,10,11,12,13,14,15);
}

__device__ inline float block_sum256(float v, float* red) {
  for (int o = 16; o; o >>= 1) v += __shfl_xor(v, o, 32);
  int w = threadIdx.x >> 5;
  __syncthreads();
  if ((threadIdx.x & 31) == 0) red[w] = v;
  __syncthreads();
  float t = red[0];
  for (int i = 1; i < 8; ++i) t += red[i];
  return t;
}

// ---- CDNA5 async global->LDS 16B copy (ASYNCcnt-tracked) --------------------
__device__ inline void async_ld16(unsigned lds_byte_off, const _Float16* g) {
  asm volatile("global_load_async_to_lds_b128 %0, %1, off"
               :: "v"(lds_byte_off), "v"(g) : "memory");
}
__device__ inline void wait_async3() { asm volatile("s_wait_asynccnt 0x3" ::: "memory"); }
__device__ inline void wait_async0() { asm volatile("s_wait_asynccnt 0x0" ::: "memory"); }

// ---------------------------------------------------------------------------
// Stage A: 20-bin DFT magnitudes of each (b, comp*53+ch) row of length 512
// ---------------------------------------------------------------------------
__global__ __launch_bounds__(64) void dft20_kernel(const float* __restrict__ img,
                                                   float* __restrict__ mags) {
  int blk = blockIdx.x;               // b*106 + c2
  __shared__ float xr[SPLEN];
  const float* row = img + (size_t)blk * SPLEN;
  for (int i = threadIdx.x; i < SPLEN; i += 64) xr[i] = row[i];
  __syncthreads();
  if (threadIdx.x < NFREQ) {
    int k = threadIdx.x;
    float w = 6.283185307179586f * (float)k / (float)SPLEN;
    float re = 0.f, im = 0.f;
    for (int n = 0; n < SPLEN; ++n) {
      float s, c; sincosf(w * (float)n, &s, &c);
      re += xr[n] * c;
      im -= xr[n] * s;
    }
    mags[(size_t)blk * NFREQ + k] = sqrtf(re * re + im * im);
  }
}

__global__ __launch_bounds__(128) void maxred_kernel(const float* __restrict__ mags,
                                                     float* __restrict__ mx, int B) {
  int c2 = blockIdx.x;                // 0..105
  __shared__ float red[128];
  float m = 0.f;
  int tot = B * NFREQ;
  for (int idx = threadIdx.x; idx < tot; idx += 128) {
    int b = idx / NFREQ, k = idx % NFREQ;
    m = fmaxf(m, mags[((size_t)b * 106 + c2) * NFREQ + k]);
  }
  red[threadIdx.x] = m;
  __syncthreads();
  for (int s = 64; s; s >>= 1) {
    if (threadIdx.x < s) red[threadIdx.x] = fmaxf(red[threadIdx.x], red[threadIdx.x + s]);
    __syncthreads();
  }
  if (threadIdx.x == 0) mx[c2] = red[0];
}

// freq conv1d + reshuffle into f16 A-matrix (B*53 rows, lda=32, cols16..31 = 0)
__global__ __launch_bounds__(256) void freqconv_kernel(const float* __restrict__ mags,
                                                       const float* __restrict__ mx,
                                                       const float* __restrict__ cw,
                                                       const float* __restrict__ cb,
                                                       _Float16* __restrict__ f16buf) {
  int b = blockIdx.x;
  __shared__ float inv[106];
  for (int i = threadIdx.x; i < 106; i += 256) {
    float m = mx[i];
    inv[i] = (m > 0.f) ? (1.f / m) : 1.f;
  }
  __syncthreads();
  for (int idx = threadIdx.x; idx < N_CH * 16; idx += 256) {  // zero pad cols
    int ch = idx >> 4, f = idx & 15;
    f16buf[((size_t)b * N_CH + ch) * 32 + 16 + f] = (_Float16)0.f;
  }
  for (int idx = threadIdx.x; idx < 8 * 106; idx += 256) {
    int w = idx % 106, o = idx / 106;
    float s = cb[o];
    for (int t = 0; t < 3; ++t) {
      int wp = w + t - 1;
      if (wp < 0 || wp >= 106) continue;
      float sc = inv[wp];
      const float* mrow = mags + ((size_t)b * 106 + wp) * NFREQ;
      const float* wrow = cw + t;               // cw[(o*20+i)*3+t]
      for (int i = 0; i < NFREQ; ++i)
        s += mrow[i] * sc * wrow[(o * NFREQ + i) * 3];
    }
    int ch = w >> 1;
    int f  = (w & 1) * 8 + o;
    f16buf[((size_t)b * N_CH + ch) * 32 + f] = (_Float16)s;
  }
}

// multi-scale temporal convs (stride 4) -> f16 A-matrix (B*53 rows, 3072 cols)
__global__ __launch_bounds__(128) void msconv_kernel(const float* __restrict__ img,
                                                     const float* __restrict__ w,
                                                     const float* __restrict__ bias,
                                                     _Float16* __restrict__ out,
                                                     int ksz, int scale) {
  int blk = blockIdx.x;               // b*53 + ch
  int b = blk / N_CH, ch = blk % N_CH;
  __shared__ float row[2 * SPLEN];
  __shared__ float wl[2 * 8 * 50];
  for (int i = threadIdx.x; i < 2 * SPLEN; i += 128) {
    int comp = i >> 9, pos = i & 511;
    row[i] = img[(((size_t)b * 2 + comp) * N_CH + ch) * SPLEN + pos];
  }
  for (int i = threadIdx.x; i < 16 * ksz; i += 128) wl[i] = w[i];
  __syncthreads();
  int wo = threadIdx.x;               // 0..127
  int pad = (ksz - 1) / 2;
  for (int o = 0; o < 8; ++o) {
    float s = bias[o];
    for (int ic = 0; ic < 2; ++ic) {
      const float* r = row + ic * SPLEN;
      const float* ww = wl + (o * 2 + ic) * ksz;
      int base = wo * 4 - pad;
      for (int t = 0; t < ksz; ++t) {
        int pos = base + t;
        if ((unsigned)pos < (unsigned)SPLEN) s += r[pos] * ww[t];
      }
    }
    out[(size_t)blk * 3072 + (scale * 8 + o) * 128 + wo] = (_Float16)s;
  }
}

// ---------------------------------------------------------------------------
// Weight prep: W[K][N] fp32  ->  Wt[N][ldt] f16 (transposed, zero-padded K)
// ---------------------------------------------------------------------------
__global__ __launch_bounds__(256) void transpose_f16(const float* __restrict__ W,
                                                     int K, int N,
                                                     _Float16* __restrict__ Wt, int ldt) {
  __shared__ float tile[32][33];
  int kb = blockIdx.x * 32, nb = blockIdx.y * 32;
  for (int it = 0; it < 4; ++it) {
    int kr = (threadIdx.x >> 5) + it * 8;
    int nc = threadIdx.x & 31;
    int gk = kb + kr, gn = nb + nc;
    tile[kr][nc] = (gk < K && gn < N) ? W[(size_t)gk * N + gn] : 0.f;
  }
  __syncthreads();
  for (int it = 0; it < 4; ++it) {
    int nr = (threadIdx.x >> 5) + it * 8;   // n within tile
    int kc = threadIdx.x & 31;              // k within tile
    int gn = nb + nr, gk = kb + kc;
    if (gn < N && gk < ldt)
      Wt[(size_t)gn * ldt + gk] = (_Float16)tile[kc][nr];
  }
}

// ---------------------------------------------------------------------------
// f16 WMMA GEMM with double-buffered async global->LDS staging.
// A: M x K f16 (row-major, lda). Bt: N x K f16 (transposed weights, ldb).
// Grid: (M/128, N/64). 8 waves, each wave a 32x32 tile (2x2 WMMA).
// C = [gelu](A*B + bias) + resid, written fp32 or f16.
// ---------------------------------------------------------------------------
template<int OUTF16, int ACT>
__global__ __launch_bounds__(256) void gemm_wmma_async(
    const _Float16* __restrict__ A, int lda,
    const _Float16* __restrict__ Bt, int ldb,
    const float* __restrict__ bias,
    const float* __restrict__ resid,
    float* __restrict__ Cf, _Float16* __restrict__ Ch, int ldc, int K) {
  __shared__ __align__(16) _Float16 Ash[2][128 * 40];   // [m][k] stride 40 halves (80B)
  __shared__ __align__(16) _Float16 Bsh[2][64 * 40];    // [n][k]
  int tid  = threadIdx.x;
  int wave = tid >> 5, lane = tid & 31;
  int sub  = lane >> 4, ln16 = lane & 15;
  int wr = wave & 3, wc = wave >> 2;                    // 4 (M) x 2 (N) waves
  int bm = blockIdx.x * 128, bn = blockIdx.y * 64;

  // per-thread async chunk coordinates (3 x 16B per tile)
  int ar0 = tid >> 2,          aq0 = tid & 3;           // A chunk 0 (rows 0..63)
  int ar1 = (tid + 256) >> 2,  aq1 = tid & 3;           // A chunk 1 (rows 64..127)
  int brn = tid >> 2,          bqn = tid & 3;           // B chunk   (rows 0..63)
  const _Float16* gA0 = A  + (size_t)(bm + ar0) * lda + aq0 * 8;
  const _Float16* gA1 = A  + (size_t)(bm + ar1) * lda + aq1 * 8;
  const _Float16* gB  = Bt + (size_t)(bn + brn) * ldb + bqn * 8;
  unsigned lA0[2], lA1[2], lB[2];
  for (int p = 0; p < 2; ++p) {
    lA0[p] = (unsigned)(size_t)&Ash[p][ar0 * 40 + aq0 * 8];
    lA1[p] = (unsigned)(size_t)&Ash[p][ar1 * 40 + aq1 * 8];
    lB[p]  = (unsigned)(size_t)&Bsh[p][brn * 40 + bqn * 8];
  }

  v8f acc[2][2] = {};
  int nT = K >> 5;

  async_ld16(lA0[0], gA0);
  async_ld16(lA1[0], gA1);
  async_ld16(lB[0],  gB);

  for (int t = 0; t < nT; ++t) {
    int cur = t & 1;
    if (t + 1 < nT) {
      int k0 = (t + 1) << 5;
      async_ld16(lA0[cur ^ 1], gA0 + k0);
      async_ld16(lA1[cur ^ 1], gA1 + k0);
      async_ld16(lB[cur ^ 1],  gB + k0);
      wait_async3();
    } else {
      wait_async0();
    }
    __syncthreads();

    const _Float16* As = Ash[cur];
    const _Float16* Bs = Bsh[cur];
    v16h af[2], bf[2];
    for (int mi = 0; mi < 2; ++mi) {
      int m = wr * 32 + mi * 16 + ln16;
      v8h lo = *(const v8h*)&As[m * 40 + sub * 8];        // K {0..7}/{8..15}
      v8h hi = *(const v8h*)&As[m * 40 + 16 + sub * 8];   // K {16..23}/{24..31}
      af[mi] = cat8(lo, hi);
    }
    for (int ni = 0; ni < 2; ++ni) {
      int n = wc * 32 + ni * 16 + ln16;
      v8h lo = *(const v8h*)&Bs[n * 40 + sub * 8];        // K {0..7}/{8..15}
      v8h hi = *(const v8h*)&Bs[n * 40 + 16 + sub * 8];   // K {16..23}/{24..31}
      bf[ni] = cat8(lo, hi);
    }
    for (int mi = 0; mi < 2; ++mi)
      for (int ni = 0; ni < 2; ++ni)
        acc[mi][ni] = __builtin_amdgcn_wmma_f32_16x16x32_f16(
            false, af[mi], false, bf[ni], (short)0, acc[mi][ni], false, false);
    __syncthreads();
  }

  for (int mi = 0; mi < 2; ++mi) {
    for (int ni = 0; ni < 2; ++ni) {
      int gcol = bn + wc * 32 + ni * 16 + ln16;
      float bv = bias ? bias[gcol] : 0.f;
      for (int r = 0; r < 8; ++r) {
        int grow = bm + wr * 32 + mi * 16 + r + 8 * sub;
        float v = acc[mi][ni][r] + bv;
        if (ACT) v = gelu_exact(v);
        if (resid) v += resid[(size_t)grow * ldc + gcol];
        if (OUTF16) Ch[(size_t)grow * ldc + gcol] = (_Float16)v;
        else        Cf[(size_t)grow * ldc + gcol] = v;
      }
    }
  }
}

// ---------------------------------------------------------------------------
// LayerNorm over rows (vectorized); OUT = float or _Float16
// ---------------------------------------------------------------------------
template<typename OUT>
__global__ __launch_bounds__(256) void ln_rows(const float* __restrict__ X,
                                               OUT* __restrict__ Y,
                                               const float* __restrict__ g,
                                               const float* __restrict__ be, int N) {
  __shared__ float red[8];
  int row = blockIdx.x;
  const float4* x4 = (const float4*)(X + (size_t)row * N);
  OUT* y = Y + (size_t)row * N;
  int n4 = N >> 2;
  float s = 0.f;
  for (int i = threadIdx.x; i < n4; i += 256) {
    float4 t = x4[i];
    s += t.x + t.y + t.z + t.w;
  }
  s = block_sum256(s, red);
  float mu = s / (float)N;
  float v = 0.f;
  for (int i = threadIdx.x; i < n4; i += 256) {
    float4 t = x4[i];
    float a = t.x - mu, b = t.y - mu, c = t.z - mu, d = t.w - mu;
    v += a * a + b * b + c * c + d * d;
  }
  v = block_sum256(v, red);
  float rstd = rsqrtf(v / (float)N + 1e-5f);
  const float4* g4 = (const float4*)g;
  const float4* b4 = (const float4*)be;
  for (int i = threadIdx.x; i < n4; i += 256) {
    float4 t = x4[i], gg = g4[i], bb = b4[i];
    float o0 = (t.x - mu) * rstd * gg.x + bb.x;
    float o1 = (t.y - mu) * rstd * gg.y + bb.y;
    float o2 = (t.z - mu) * rstd * gg.z + bb.z;
    float o3 = (t.w - mu) * rstd * gg.w + bb.w;
    if (sizeof(OUT) == 2) {
      v4h o = { (_Float16)o0, (_Float16)o1, (_Float16)o2, (_Float16)o3 };
      *(v4h*)&y[i * 4] = o;
    } else {
      float4 o = { o0, o1, o2, o3 };
      *(float4*)&y[i * 4] = o;
    }
  }
}

// ---------------------------------------------------------------------------
// Token assembly: x = [region_tokens ; concat(freq_feat, time_feat)] + pos
// ---------------------------------------------------------------------------
__global__ __launch_bounds__(256) void assemble_kernel(const float* __restrict__ tf,
                                                       const float* __restrict__ tt,
                                                       const float* __restrict__ region,
                                                       const float* __restrict__ pos,
                                                       float* __restrict__ x) {
  int blk = blockIdx.x;               // b*60 + tok
  int b = blk / N_TOK, tok = blk % N_TOK;
  float* xr = x + (size_t)blk * FD;
  const float* pr = pos + (size_t)tok * FD;
  for (int d = threadIdx.x; d < FD; d += 256) {
    float v;
    if (tok < N_REG) v = region[(size_t)tok * FD + d];
    else {
      int ch = tok - N_REG;
      v = (d < DM) ? tf[((size_t)b * N_CH + ch) * DM + d]
                   : tt[((size_t)b * N_CH + ch) * DM + (d - DM)];
    }
    xr[d] = v + pr[d];
  }
}

// ---------------------------------------------------------------------------
// Attention: one workgroup per (b, head). QK^T and PV via WMMA on f16 qkv.
// feat_att bias dropped: per-row constant cancels in softmax.
// ---------------------------------------------------------------------------
__global__ __launch_bounds__(256) void attn_wmma(const _Float16* __restrict__ qkv,
                                                 _Float16* __restrict__ o) {
  int bh = blockIdx.x;
  int b = bh >> 3, h = bh & 7;
  __shared__ __align__(16) _Float16 Qs[64 * 80];   // Q (reused for probs)
  __shared__ __align__(16) _Float16 Ks[64 * 80];   // K rows (B^T for QK^T)
  __shared__ __align__(16) _Float16 Vt[64 * 80];   // V transposed: [d][j]
  __shared__ float Ds[64 * 68];
  __shared__ unsigned long long Msk[64];

  int tid  = threadIdx.x;
  int wave = tid >> 5, lane = tid & 31;
  int sub  = lane >> 4, ln16 = lane & 15;
  const _Float16* base = qkv + (size_t)b * N_TOK * 1536;

  if (tid < 64) Msk[tid] = MASK.row[tid];
  for (int e = tid; e < 512; e += 256) {           // 64 rows x 8 chunks of v8h
    int r = e >> 3, q = e & 7;
    v8h qz = {}, kz = {};
    if (r < N_TOK) {
      qz = *(const v8h*)&base[(size_t)r * 1536 + h * DH + q * 8];
      kz = *(const v8h*)&base[(size_t)r * 1536 + DM + h * DH + q * 8];
    }
    *(v8h*)&Qs[r * 80 + q * 8] = qz;
    *(v8h*)&Ks[r * 80 + q * 8] = kz;
  }
  for (int e = tid; e < 512; e += 256) {           // V with transpose
    int j = e >> 3, q = e & 7;
    v8h vz = {};
    if (j < N_TOK) vz = *(const v8h*)&base[(size_t)j * 1536 + 2 * DM + h * DH + q * 8];
    for (int u = 0; u < 8; ++u) Vt[(q * 8 + u) * 80 + j] = vz[u];
  }
  __syncthreads();

  // dots = (Q K^T) * 1/8, masked into Ds
  for (int t = wave; t < 16; t += 8) {
    int ti = t >> 2, tj = t & 3;
    v8f acc = {};
    for (int kc = 0; kc < DH; kc += 32) {
      int m = ti * 16 + ln16;
      v8h a0 = *(const v8h*)&Qs[m * 80 + kc + sub * 8];
      v8h a1 = *(const v8h*)&Qs[m * 80 + kc + 16 + sub * 8];
      int n = tj * 16 + ln16;
      v8h b0 = *(const v8h*)&Ks[n * 80 + kc + sub * 8];
      v8h b1 = *(const v8h*)&Ks[n * 80 + kc + 16 + sub * 8];
      acc = __builtin_amdgcn_wmma_f32_16x16x32_f16(
          false, cat8(a0, a1), false, cat8(b0, b1), (short)0, acc, false, false);
    }
    int j = tj * 16 + ln16;
    for (int r = 0; r < 8; ++r) {
      int i = ti * 16 + r + 8 * sub;
      unsigned long long mrow = Msk[i];
      bool ok = (mrow >> j) & 1ull;
      Ds[i * 68 + j] = ok ? acc[r] * 0.125f : -1e30f;
    }
  }
  __syncthreads();

  // parallel softmax: 4 lanes per row, 16 columns each
  {
    int i = tid >> 2, q4 = tid & 3;
    float* drow = Ds + i * 68 + q4 * 16;
    float m = -3.0e38f;
    for (int jj = 0; jj < 16; ++jj) m = fmaxf(m, drow[jj]);
    m = fmaxf(m, __shfl_xor(m, 1, 4));
    m = fmaxf(m, __shfl_xor(m, 2, 4));
    float ssum = 0.f;
    float e[16];
    for (int jj = 0; jj < 16; ++jj) { e[jj] = expf(drow[jj] - m); ssum += e[jj]; }
    ssum += __shfl_xor(ssum, 1, 4);
    ssum += __shfl_xor(ssum, 2, 4);
    float inv = 1.f / ssum;
    _Float16* prow = Qs + i * 80 + q4 * 16;
    for (int jj = 0; jj < 16; ++jj) prow[jj] = (_Float16)(e[jj] * inv);
  }
  __syncthreads();

  // out = P @ V
  for (int t = wave; t < 16; t += 8) {
    int ti = t >> 2, td = t & 3;
    v8f acc = {};
    for (int kc = 0; kc < 64; kc += 32) {
      int m = ti * 16 + ln16;
      v8h a0 = *(const v8h*)&Qs[m * 80 + kc + sub * 8];
      v8h a1 = *(const v8h*)&Qs[m * 80 + kc + 16 + sub * 8];
      int n = td * 16 + ln16;
      v8h b0 = *(const v8h*)&Vt[n * 80 + kc + sub * 8];
      v8h b1 = *(const v8h*)&Vt[n * 80 + kc + 16 + sub * 8];
      acc = __builtin_amdgcn_wmma_f32_16x16x32_f16(
          false, cat8(a0, a1), false, cat8(b0, b1), (short)0, acc, false, false);
    }
    for (int r = 0; r < 8; ++r) {
      int i = ti * 16 + r + 8 * sub;
      if (i < N_TOK) {
        int d = td * 16 + ln16;
        o[((size_t)b * N_TOK + i) * DM + h * DH + d] = (_Float16)acc[r];
      }
    }
  }
}

// ---------------------------------------------------------------------------
// Head: out[b] = gelu(LN(x[b,0,:])) @ head_w + head_b
// ---------------------------------------------------------------------------
__global__ __launch_bounds__(256) void head_kernel(const float* __restrict__ x,
                                                   const float* __restrict__ g,
                                                   const float* __restrict__ be,
                                                   const float* __restrict__ hw,
                                                   const float* __restrict__ hb,
                                                   float* __restrict__ out) {
  __shared__ float red[8];
  int b = blockIdx.x;
  const float* r = x + (size_t)b * N_TOK * FD;  // token 0 row
  float s = 0.f;
  for (int i = threadIdx.x; i < FD; i += 256) s += r[i];
  s = block_sum256(s, red);
  float mu = s * (1.f / (float)FD);
  float v = 0.f;
  for (int i = threadIdx.x; i < FD; i += 256) { float d = r[i] - mu; v += d * d; }
  v = block_sum256(v, red);
  float rstd = rsqrtf(v * (1.f / (float)FD) + 1e-5f);
  float d0 = 0.f, d1 = 0.f;
  for (int i = threadIdx.x; i < FD; i += 256) {
    float hv = gelu_exact((r[i] - mu) * rstd * g[i] + be[i]);
    d0 += hv * hw[i * 2];
    d1 += hv * hw[i * 2 + 1];
  }
  d0 = block_sum256(d0, red);
  d1 = block_sum256(d1, red);
  if (threadIdx.x == 0) {
    out[b * 2]     = d0 + hb[0];
    out[b * 2 + 1] = d1 + hb[1];
  }
}

// ---------------------------------------------------------------------------
// Host orchestration
// ---------------------------------------------------------------------------
#define PIN(i) ((const float*)d_in[(i)])

extern "C" void kernel_launch(void* const* d_in, const int* in_sizes, int n_in,
                              void* d_out, int out_size, void* d_ws, size_t ws_size,
                              hipStream_t stream) {
  (void)n_in; (void)out_size; (void)ws_size;
  const float* img = PIN(0);
  const int B    = in_sizes[0] / (2 * N_CH * SPLEN);
  const int Mtok = B * N_TOK;                       // 3840 (mult of 128)
  const int Mch  = B * N_CH;                        // 3392
  const int Mchp = (Mch + 127) & ~127;              // 3456 padded rows

  const int LBASE = 23, LSTRIDE = 16;

  char* wsb = (char*)d_ws;
  size_t off = 0;
  auto carve = [&](size_t bytes) { char* p = wsb + off; off += (bytes + 255) & ~(size_t)255; return p; };

  float*     f_mags = (float*)carve((size_t)B * 106 * NFREQ * 4);
  float*     f_mx   = (float*)carve(512);
  _Float16*  f_f16  = (_Float16*)carve((size_t)Mchp * 32 * 2);
  float*     f_tf   = (float*)carve((size_t)Mchp * DM * 4);
  float*     f_tt   = (float*)carve((size_t)Mchp * DM * 4);
  _Float16*  f_ms   = (_Float16*)carve((size_t)Mchp * 3072 * 2);
  float*     f_x    = (float*)carve((size_t)Mtok * FD * 4);
  _Float16*  f_xn   = (_Float16*)carve((size_t)Mtok * FD * 2);
  _Float16*  f_qkv  = (_Float16*)carve((size_t)Mtok * 1536 * 2);
  _Float16*  f_ao   = (_Float16*)carve((size_t)Mtok * DM * 2);
  _Float16*  f_ff   = (_Float16*)carve((size_t)Mtok * MLPD * 2);
  _Float16*  w_freq = (_Float16*)carve((size_t)DM * 32 * 2);        // 512 x 32
  _Float16*  w_ms   = (_Float16*)carve((size_t)DM * 3072 * 2);      // 512 x 3072
  _Float16*  w_qkv  = (_Float16*)carve((size_t)1536 * FD * 2);      // 1536 x 1024
  _Float16*  w_out  = (_Float16*)carve((size_t)FD * DM * 2);        // 1024 x 512
  _Float16*  w_ff1  = (_Float16*)carve((size_t)MLPD * FD * 2);      // 4096 x 1024
  _Float16*  w_ff2  = (_Float16*)carve((size_t)FD * MLPD * 2);      // 1024 x 4096

  // ---- frequency branch ----
  dft20_kernel<<<B * 106, 64, 0, stream>>>(img, f_mags);
  maxred_kernel<<<106, 128, 0, stream>>>(f_mags, f_mx, B);
  freqconv_kernel<<<B, 256, 0, stream>>>(f_mags, f_mx, PIN(1), PIN(2), f_f16);
  transpose_f16<<<dim3(1, 16), 256, 0, stream>>>(PIN(3), 16, DM, w_freq, 32);
  gemm_wmma_async<0, 0><<<dim3(Mchp / 128, DM / 64), 256, 0, stream>>>(
      f_f16, 32, w_freq, 32, PIN(4), nullptr, f_tf, nullptr, DM, 32);
  ln_rows<float><<<Mch, 256, 0, stream>>>(f_tf, f_tf, PIN(5), PIN(6), DM);

  // ---- multi-scale temporal branch ----
  const int KS[3] = {50, 25, 12};
  for (int s = 0; s < 3; ++s)
    msconv_kernel<<<Mch, 128, 0, stream>>>(img, PIN(7 + s), PIN(10 + s), f_ms, KS[s], s);
  transpose_f16<<<dim3(96, 16), 256, 0, stream>>>(PIN(13), 3072, DM, w_ms, 3072);
  gemm_wmma_async<0, 0><<<dim3(Mchp / 128, DM / 64), 256, 0, stream>>>(
      f_ms, 3072, w_ms, 3072, PIN(14), nullptr, f_tt, nullptr, DM, 3072);
  ln_rows<float><<<Mch, 256, 0, stream>>>(f_tt, f_tt, PIN(15), PIN(16), DM);

  // ---- assemble tokens ----
  assemble_kernel<<<Mtok, 256, 0, stream>>>(f_tf, f_tt, PIN(17), PIN(18), f_x);

  // ---- transformer layers ----
  for (int L = 0; L < 6; ++L) {
    int pb = LBASE + L * LSTRIDE;
    transpose_f16<<<dim3(32, 48), 256, 0, stream>>>(PIN(pb + 2), FD, 1536, w_qkv, FD);
    transpose_f16<<<dim3(16, 32), 256, 0, stream>>>(PIN(pb + 3), DM, FD, w_out, DM);
    transpose_f16<<<dim3(32, 128), 256, 0, stream>>>(PIN(pb + 12), FD, MLPD, w_ff1, FD);
    transpose_f16<<<dim3(128, 32), 256, 0, stream>>>(PIN(pb + 14), MLPD, FD, w_ff2, MLPD);

    ln_rows<_Float16><<<Mtok, 256, 0, stream>>>(f_x, f_xn, PIN(pb + 0), PIN(pb + 1), FD);
    gemm_wmma_async<1, 0><<<dim3(Mtok / 128, 1536 / 64), 256, 0, stream>>>(
        f_xn, FD, w_qkv, FD, nullptr, nullptr, nullptr, f_qkv, 1536, FD);
    attn_wmma<<<B * HEADS, 256, 0, stream>>>(f_qkv, f_ao);
    gemm_wmma_async<0, 0><<<dim3(Mtok / 128, FD / 64), 256, 0, stream>>>(
        f_ao, DM, w_out, DM, PIN(pb + 4), f_x, f_x, nullptr, FD, DM);
    ln_rows<_Float16><<<Mtok, 256, 0, stream>>>(f_x, f_xn, PIN(pb + 10), PIN(pb + 11), FD);
    gemm_wmma_async<1, 1><<<dim3(Mtok / 128, MLPD / 64), 256, 0, stream>>>(
        f_xn, FD, w_ff1, FD, PIN(pb + 13), nullptr, nullptr, f_ff, MLPD, FD);
    gemm_wmma_async<0, 0><<<dim3(Mtok / 128, FD / 64), 256, 0, stream>>>(
        f_ff, MLPD, w_ff2, MLPD, PIN(pb + 15), f_x, f_x, nullptr, FD, MLPD);
  }

  // ---- head ----
  head_kernel<<<B, 256, 0, stream>>>(f_x, PIN(19), PIN(20), PIN(21), PIN(22), (float*)d_out);
}
